// RoutedQwenDecoderLayer_29987461661147
// MI455X (gfx1250) — matrix operand order repaired and checked by reading, not compiled, not run on hardware
//
#include <hip/hip_runtime.h>
#include <hip/hip_bf16.h>

// ---------------------------------------------------------------------------
// Problem constants (match the reference)
// ---------------------------------------------------------------------------
constexpr int B_  = 4;
constexpr int P_  = 2048;
constexpr int D_  = 1024;
constexpr int H_  = 16;
constexpr int KV_ = 8;
constexpr int HD_ = 64;
constexpr int F_  = 4096;
constexpr int E_  = 16;
constexpr int FE_ = 256;
constexpr int M_  = B_ * P_;          // 8192 tokens

typedef _Float16 f16;
typedef __attribute__((ext_vector_type(8)))  _Float16 v8h;
typedef __attribute__((ext_vector_type(16))) _Float16 v16h;
typedef __attribute__((ext_vector_type(8)))  float    v8f;
typedef __attribute__((__vector_size__(4 * sizeof(int)))) int v4i;   // b128 payload

#define AS3 __attribute__((address_space(3)))
#define AS1 __attribute__((address_space(1)))

// CDNA5 async global->LDS copy (bypasses VGPRs, tracked by ASYNCcnt)
#if __has_builtin(__builtin_amdgcn_global_load_async_to_lds_b128)
#define USE_ASYNC_LDS 1
#else
#define USE_ASYNC_LDS 0
#endif

__device__ __forceinline__ void wait_async0() {
#if __has_builtin(__builtin_amdgcn_s_wait_asynccnt)
    __builtin_amdgcn_s_wait_asynccnt(0);
#else
    asm volatile("s_wait_asynccnt 0x0" ::: "memory");
#endif
}

__device__ __forceinline__ v8f wmma32(v16h a, v16h b, v8f c) {
    // D = A(16x32 f16) * B(32x16 f16) + C(16x16 f32)
    return __builtin_amdgcn_wmma_f32_16x16x32_f16(
        /*neg_a=*/false, a, /*neg_b=*/false, b,
        /*c_mod=*/(short)0, c, /*reuse_a=*/false, /*reuse_b=*/false);
}

__device__ __forceinline__ v16h join16(v8h lo, v8h hi) {
    return __builtin_shufflevector(lo, hi, 0,1,2,3,4,5,6,7,8,9,10,11,12,13,14,15);
}

// ---------------------------------------------------------------------------
// Generic WMMA GEMM:  C[M,N] = A[M,K] (f16, row-major) * B[K,N] (f16, row-major)
// mode 0: fp32 out   mode 1: f16 out   mode 2: f16 out with SiLU
// M % 128 == 0, N % 128 == 0, K % 32 == 0 (true for all shapes here)
// ---------------------------------------------------------------------------
#define TILE_M 128
#define TILE_N 128
#define TILE_K 32

__global__ __launch_bounds__(256)
void gemm_wmma_kernel(const f16* __restrict__ A, const f16* __restrict__ Bw,
                      void* __restrict__ Cout, int Md, int Nd, int Kd, int mode)
{
    __shared__ f16 As[TILE_M * TILE_K];   // [m][k]
    __shared__ f16 Bs[TILE_N * TILE_K];   // [n][k]  (transposed while staging)

    const int tid     = threadIdx.x;
    const int lane    = tid & 31;
    const int wave    = tid >> 5;
    const int half_id = lane >> 4;
    const int l15     = lane & 15;

    const int row0 = blockIdx.y * TILE_M;
    const int col0 = blockIdx.x * TILE_N;
    const int wm   = wave & 1;      // 2 waves along M -> 64 rows each
    const int wn   = wave >> 1;     // 4 waves along N -> 32 cols each

    v8f acc[4][2];
    #pragma unroll
    for (int i = 0; i < 4; ++i)
        #pragma unroll
        for (int j = 0; j < 2; ++j)
            acc[i][j] = {};

    const int ar = tid >> 1;            // A stage: row 0..127
    const int ak = (tid & 1) * 16;      //          k-offset 0/16
    const int bk = tid >> 3;            // B stage: k row 0..31
    const int bn = (tid & 7) * 16;      //          n-offset 0..112

    for (int kb = 0; kb < Kd; kb += TILE_K) {
        // ---- stage A tile: async global->LDS (CDNA5 async path, ASYNCcnt) ----
        const f16* ag = A + (size_t)(row0 + ar) * Kd + kb + ak;
        f16* asl = As + ar * TILE_K + ak;
#if USE_ASYNC_LDS
        __builtin_amdgcn_global_load_async_to_lds_b128(
            (AS1 v4i*)ag,       (AS3 v4i*)asl,       0, 0);
        __builtin_amdgcn_global_load_async_to_lds_b128(
            (AS1 v4i*)(ag + 8), (AS3 v4i*)(asl + 8), 0, 0);
#else
        {
            v8h a0 = *(const v8h*)(ag);
            v8h a1 = *(const v8h*)(ag + 8);
            *(v8h*)(asl)     = a0;
            *(v8h*)(asl + 8) = a1;
        }
#endif

        // ---- stage B tile, transpose into [n][k] ----
        const f16* bg = Bw + (size_t)(kb + bk) * Nd + col0 + bn;
        v8h b0 = *(const v8h*)(bg);
        v8h b1 = *(const v8h*)(bg + 8);
        #pragma unroll
        for (int j = 0; j < 8; ++j) Bs[(bn + j)     * TILE_K + bk] = b0[j];
        #pragma unroll
        for (int j = 0; j < 8; ++j) Bs[(bn + 8 + j) * TILE_K + bk] = b1[j];

        if (kb + TILE_K < Kd) {
            __builtin_prefetch(ag + TILE_K, 0, 1);                 // global_prefetch_b8
            __builtin_prefetch(bg + (size_t)TILE_K * Nd, 0, 1);
        }
#if USE_ASYNC_LDS
        wait_async0();
#endif
        __syncthreads();

        // ---- fragments (documented 16-bit WMMA VGPR layouts) ----
        v16h afrag[4];
        #pragma unroll
        for (int i = 0; i < 4; ++i) {
            const f16* p = As + (wm * 64 + i * 16 + l15) * TILE_K + 8 * half_id;
            afrag[i] = join16(*(const v8h*)p, *(const v8h*)(p + 16));
        }
        v16h bfrag[2];
        #pragma unroll
        for (int j = 0; j < 2; ++j) {
            const f16* p = Bs + (wn * 32 + j * 16 + l15) * TILE_K + 8 * half_id;
            bfrag[j] = join16(*(const v8h*)p, *(const v8h*)(p + 16));
        }
        #pragma unroll
        for (int i = 0; i < 4; ++i)
            #pragma unroll
            for (int j = 0; j < 2; ++j)
                acc[i][j] = wmma32(afrag[i], bfrag[j], acc[i][j]);
        __syncthreads();
    }

    // ---- epilogue (C layout: lane -> col, vgpr r -> row r + 8*half) ----
    #pragma unroll
    for (int i = 0; i < 4; ++i) {
        #pragma unroll
        for (int j = 0; j < 2; ++j) {
            const int cc = col0 + wn * 32 + j * 16 + l15;
            #pragma unroll
            for (int r = 0; r < 8; ++r) {
                const int rr = row0 + wm * 64 + i * 16 + r + 8 * half_id;
                const float v = acc[i][j][r];
                const size_t idx = (size_t)rr * Nd + cc;
                if (mode == 0) {
                    ((float*)Cout)[idx] = v;
                } else if (mode == 1) {
                    ((f16*)Cout)[idx] = (f16)v;
                } else {
                    ((f16*)Cout)[idx] = (f16)(v / (1.0f + __expf(-v)));   // SiLU
                }
            }
        }
    }
}

// ---------------------------------------------------------------------------
// RMSNorm: one block per row (row length n = 1024), f16 output
// ---------------------------------------------------------------------------
__global__ __launch_bounds__(256)
void rmsnorm_kernel(const float* __restrict__ x, const float* __restrict__ w,
                    f16* __restrict__ out, int n)
{
    __shared__ float red[8];
    const size_t row = blockIdx.x;
    const float* xr = x + row * n;
    float ss = 0.f;
    for (int i = threadIdx.x; i < n; i += 256) { float v = xr[i]; ss += v * v; }
    #pragma unroll
    for (int off = 16; off; off >>= 1) ss += __shfl_xor(ss, off, 32);
    if ((threadIdx.x & 31) == 0) red[threadIdx.x >> 5] = ss;
    __syncthreads();
    if (threadIdx.x < 8) {
        float v = red[threadIdx.x];
        #pragma unroll
        for (int off = 4; off; off >>= 1) v += __shfl_xor(v, off, 32);
        if (threadIdx.x == 0) red[0] = v;
    }
    __syncthreads();
    const float inv = rsqrtf(red[0] / (float)n + 1e-6f);
    for (int i = threadIdx.x; i < n; i += 256)
        out[row * n + i] = (f16)(xr[i] * inv * w[i]);
}

// residual add + RMSNorm (x1 = x + o ; h2 = rmsnorm(x1) * w)
__global__ __launch_bounds__(256)
void resid_rmsnorm_kernel(const float* __restrict__ x, const float* __restrict__ o,
                          const float* __restrict__ w, float* __restrict__ x1,
                          f16* __restrict__ out, int n)
{
    __shared__ float red[8];
    const size_t row = blockIdx.x;
    float ss = 0.f;
    for (int i = threadIdx.x; i < n; i += 256) {
        const float v = x[row * n + i] + o[row * n + i];
        x1[row * n + i] = v;
        ss += v * v;
    }
    #pragma unroll
    for (int off = 16; off; off >>= 1) ss += __shfl_xor(ss, off, 32);
    if ((threadIdx.x & 31) == 0) red[threadIdx.x >> 5] = ss;
    __syncthreads();
    if (threadIdx.x < 8) {
        float v = red[threadIdx.x];
        #pragma unroll
        for (int off = 4; off; off >>= 1) v += __shfl_xor(v, off, 32);
        if (threadIdx.x == 0) red[0] = v;
    }
    __syncthreads();
    const float inv = rsqrtf(red[0] / (float)n + 1e-6f);
    for (int i = threadIdx.x; i < n; i += 256)
        out[row * n + i] = (f16)(x1[row * n + i] * inv * w[i]);
}

// ---------------------------------------------------------------------------
// RoPE + bias + pack to [b][h][p][hd] f16 (Q/K). One block per token.
// ---------------------------------------------------------------------------
__global__ __launch_bounds__(256)
void rope_pack_kernel(const float* __restrict__ src, const float* __restrict__ bias,
                      f16* __restrict__ dst, int nheads)
{
    const int m = blockIdx.x;
    const int b = m / P_, p = m % P_;
    const int rowlen = nheads * HD_;
    for (int idx = threadIdx.x; idx < nheads * (HD_ / 2); idx += 256) {
        const int h = idx >> 5;          // HD_/2 == 32
        const int d = idx & 31;
        const float inv = __powf(10000.0f, -(float)(2 * d) / (float)HD_);
        float s, c;
        __sincosf((float)p * inv, &s, &c);
        const float v1 = src[(size_t)m * rowlen + h * HD_ + d]      + bias[h * HD_ + d];
        const float v2 = src[(size_t)m * rowlen + h * HD_ + d + 32] + bias[h * HD_ + d + 32];
        f16* dp = dst + ((size_t)(b * nheads + h) * P_ + p) * HD_;
        dp[d]      = (f16)(v1 * c - v2 * s);
        dp[d + 32] = (f16)(v2 * c + v1 * s);
    }
}

// V: bias + pack transposed to [b][kv][hd][P] f16 (so PV B-fragments are contiguous)
__global__ __launch_bounds__(256)
void pack_v_kernel(const float* __restrict__ src, const float* __restrict__ bias,
                   f16* __restrict__ dst)
{
    const int m = blockIdx.x;
    const int b = m / P_, p = m % P_;
    for (int idx = threadIdx.x; idx < KV_ * HD_; idx += 256) {
        const int kv = idx >> 6, d = idx & 63;
        dst[((size_t)(b * KV_ + kv) * HD_ + d) * P_ + p] =
            (f16)(src[(size_t)m * (KV_ * HD_) + idx] + bias[idx]);
    }
}

// ---------------------------------------------------------------------------
// Flash attention: one wave per (16 queries, head, batch). WMMA QK^T and PV.
// q: [b][h][P][64]  k: [b][kv][P][64]  vt: [b][kv][64][P]  out: [tok][H*64] f16
// ---------------------------------------------------------------------------
__global__ __launch_bounds__(32)
void attn_kernel(const f16* __restrict__ q, const f16* __restrict__ k,
                 const f16* __restrict__ vt, f16* __restrict__ out)
{
    __shared__ f16 Pl[16 * 32];
    const int qb = blockIdx.x, h = blockIdx.y, b = blockIdx.z;
    const int kv = h >> 1;                     // H/KV == 2
    const int lane = threadIdx.x & 31;
    const int half_id = lane >> 4, l15 = lane & 15;

    // Q fragments (A-matrix, 16x64 over two K=32 chunks)
    const f16* qp = q + ((size_t)(b * H_ + h) * P_ + qb * 16 + l15) * HD_;
    v16h qa[2];
    #pragma unroll
    for (int c = 0; c < 2; ++c) {
        const f16* p = qp + c * 32 + 8 * half_id;
        qa[c] = join16(*(const v8h*)p, *(const v8h*)(p + 16));
    }

    v8f oacc[4];
    #pragma unroll
    for (int t = 0; t < 4; ++t) oacc[t] = {};
    float mrow[8], lrow[8];
    #pragma unroll
    for (int r = 0; r < 8; ++r) { mrow[r] = -1e30f; lrow[r] = 0.f; }

    const int nkc = (qb * 16 + 15) / 32 + 1;   // causal chunk bound
    const f16* kbase = k  + (size_t)(b * KV_ + kv) * P_ * HD_;
    const f16* vbase = vt + (size_t)(b * KV_ + kv) * HD_ * P_;

    for (int kc = 0; kc < nkc; ++kc) {
        float sv[2][8];
        #pragma unroll
        for (int t = 0; t < 2; ++t) {
            const int key = kc * 32 + t * 16 + l15;
            const f16* kp = kbase + (size_t)key * HD_;
            v8f s = {};
            #pragma unroll
            for (int c = 0; c < 2; ++c) {
                const f16* p = kp + c * 32 + 8 * half_id;
                v16h bf = join16(*(const v8h*)p, *(const v8h*)(p + 16));
                s = wmma32(qa[c], bf, s);
            }
            #pragma unroll
            for (int r = 0; r < 8; ++r) {
                const int qrow = qb * 16 + r + 8 * half_id;
                sv[t][r] = (key <= qrow) ? s[r] * 0.125f : -1.0e9f;   // 1/sqrt(64)
            }
        }
        // online softmax
        float alpha[8];
        #pragma unroll
        for (int r = 0; r < 8; ++r) {
            float mx = fmaxf(sv[0][r], sv[1][r]);
            #pragma unroll
            for (int off = 1; off < 16; off <<= 1) mx = fmaxf(mx, __shfl_xor(mx, off, 32));
            const float mnew = fmaxf(mrow[r], mx);
            alpha[r] = __expf(mrow[r] - mnew);
            const float p0 = __expf(sv[0][r] - mnew);
            const float p1 = __expf(sv[1][r] - mnew);
            float ps = p0 + p1;
            #pragma unroll
            for (int off = 1; off < 16; off <<= 1) ps += __shfl_xor(ps, off, 32);
            lrow[r] = lrow[r] * alpha[r] + ps;
            mrow[r] = mnew;
            sv[0][r] = p0; sv[1][r] = p1;
        }
        #pragma unroll
        for (int t = 0; t < 4; ++t)
            #pragma unroll
            for (int r = 0; r < 8; ++r) oacc[t][r] *= alpha[r];

        // transpose P through LDS into A-fragment layout
        #pragma unroll
        for (int t = 0; t < 2; ++t)
            #pragma unroll
            for (int r = 0; r < 8; ++r)
                Pl[(r + 8 * half_id) * 32 + t * 16 + l15] = (f16)sv[t][r];
        __syncthreads();
        const f16* pp = Pl + l15 * 32 + 8 * half_id;
        v16h pa = join16(*(const v8h*)pp, *(const v8h*)(pp + 16));
        #pragma unroll
        for (int nt = 0; nt < 4; ++nt) {
            const f16* vp = vbase + (size_t)(nt * 16 + l15) * P_ + kc * 32 + 8 * half_id;
            v16h vb = join16(*(const v8h*)vp, *(const v8h*)(vp + 16));
            oacc[nt] = wmma32(pa, vb, oacc[nt]);
        }
        __syncthreads();
    }

    #pragma unroll
    for (int r = 0; r < 8; ++r) {
        const float invl = 1.f / lrow[r];
        const size_t tok = (size_t)b * P_ + qb * 16 + r + 8 * half_id;
        #pragma unroll
        for (int nt = 0; nt < 4; ++nt)
            out[tok * (H_ * HD_) + h * HD_ + nt * 16 + l15] = (f16)(oacc[nt][r] * invl);
    }
}

// ---------------------------------------------------------------------------
// Small elementwise / conversion kernels
// ---------------------------------------------------------------------------
__global__ void cvt_kernel(const float* __restrict__ in, f16* __restrict__ out, size_t n)
{
    const size_t stride = (size_t)gridDim.x * blockDim.x;
    for (size_t i = (size_t)blockIdx.x * blockDim.x + threadIdx.x; i < n; i += stride)
        out[i] = (f16)in[i];
}

// (E, D, FE) -> (D, E*FE) f16
__global__ void cvt_texp_kernel(const float* __restrict__ in, f16* __restrict__ out)
{
    const size_t n = (size_t)E_ * D_ * FE_;
    const size_t stride = (size_t)gridDim.x * blockDim.x;
    for (size_t i = (size_t)blockIdx.x * blockDim.x + threadIdx.x; i < n; i += stride) {
        const size_t e = i / ((size_t)D_ * FE_);
        const size_t rem = i % ((size_t)D_ * FE_);
        const size_t d = rem / FE_, f = rem % FE_;
        out[d * (size_t)(E_ * FE_) + e * FE_ + f] = (f16)in[i];
    }
}

__global__ void mul_kernel(const f16* __restrict__ a, const f16* __restrict__ b,
                           f16* __restrict__ c, size_t n)
{
    const size_t stride = (size_t)gridDim.x * blockDim.x;
    for (size_t i = (size_t)blockIdx.x * blockDim.x + threadIdx.x; i < n; i += stride)
        c[i] = (f16)((float)a[i] * (float)b[i]);
}

// act[m][e*FE+f] = g*u*mask[m][e]
__global__ void mask_mul_kernel(const f16* __restrict__ g, const f16* __restrict__ u,
                                const float* __restrict__ mask, f16* __restrict__ c)
{
    const size_t n = (size_t)M_ * F_;
    const size_t stride = (size_t)gridDim.x * blockDim.x;
    for (size_t i = (size_t)blockIdx.x * blockDim.x + threadIdx.x; i < n; i += stride) {
        const size_t m = i >> 12;             // F_ == 4096
        const int    e = (int)((i & 4095) >> 8);   // FE_ == 256
        c[i] = (f16)((float)g[i] * (float)u[i] * mask[m * E_ + e]);
    }
}

__global__ void final_add_kernel(const float* __restrict__ x1, const float* __restrict__ a,
                                 const float* __restrict__ b, float* __restrict__ out, size_t n)
{
    const size_t stride = (size_t)gridDim.x * blockDim.x;
    for (size_t i = (size_t)blockIdx.x * blockDim.x + threadIdx.x; i < n; i += stride)
        out[i] = x1[i] + a[i] + b[i];
}

// ---------------------------------------------------------------------------
// Host orchestration
// ---------------------------------------------------------------------------
extern "C" void kernel_launch(void* const* d_in, const int* in_sizes, int n_in,
                              void* d_out, int out_size, void* d_ws, size_t ws_size,
                              hipStream_t stream)
{
    (void)in_sizes; (void)n_in; (void)out_size; (void)ws_size;

    const float* x      = (const float*)d_in[0];
    const float* emask  = (const float*)d_in[1];
    const float* ln1_w  = (const float*)d_in[2];
    const float* wq     = (const float*)d_in[3];
    const float* bq     = (const float*)d_in[4];
    const float* wk     = (const float*)d_in[5];
    const float* bk     = (const float*)d_in[6];
    const float* wv     = (const float*)d_in[7];
    const float* bv     = (const float*)d_in[8];
    const float* wo     = (const float*)d_in[9];
    const float* ln2_w  = (const float*)d_in[10];
    const float* w_gate = (const float*)d_in[11];
    const float* w_up   = (const float*)d_in[12];
    const float* w_down = (const float*)d_in[13];
    const float* we_g   = (const float*)d_in[14];
    const float* we_u   = (const float*)d_in[15];
    const float* we_d   = (const float*)d_in[16];
    float* out = (float*)d_out;

    // bump allocator over workspace
    char* base = (char*)d_ws;
    size_t off = 0;
    auto alloc = [&](size_t bytes) -> void* {
        void* p = base + off;
        off += (bytes + 255) & ~(size_t)255;
        return p;
    };

    // f16 weights
    f16* wq16  = (f16*)alloc((size_t)D_ * 1024 * 2);
    f16* wk16  = (f16*)alloc((size_t)D_ * 512 * 2);
    f16* wv16  = (f16*)alloc((size_t)D_ * 512 * 2);
    f16* wo16  = (f16*)alloc((size_t)1024 * D_ * 2);
    f16* wg16  = (f16*)alloc((size_t)D_ * F_ * 2);
    f16* wu16  = (f16*)alloc((size_t)D_ * F_ * 2);
    f16* wd16  = (f16*)alloc((size_t)F_ * D_ * 2);
    f16* weg16 = (f16*)alloc((size_t)D_ * F_ * 2);
    f16* weu16 = (f16*)alloc((size_t)D_ * F_ * 2);
    f16* wed16 = (f16*)alloc((size_t)F_ * D_ * 2);
    // activations
    f16*   h16    = (f16*)  alloc((size_t)M_ * D_ * 2);
    float* qf     = (float*)alloc((size_t)M_ * 1024 * 4);
    float* kf     = (float*)alloc((size_t)M_ * 512 * 4);
    float* vf     = (float*)alloc((size_t)M_ * 512 * 4);
    f16*   q16    = (f16*)  alloc((size_t)M_ * 1024 * 2);
    f16*   k16    = (f16*)  alloc((size_t)M_ * 512 * 2);
    f16*   v16t   = (f16*)  alloc((size_t)M_ * 512 * 2);
    f16*   attn16 = (f16*)  alloc((size_t)M_ * 1024 * 2);
    float* of     = (float*)alloc((size_t)M_ * D_ * 4);
    float* x1     = (float*)alloc((size_t)M_ * D_ * 4);
    f16*   h2_16  = (f16*)  alloc((size_t)M_ * D_ * 2);
    f16*   g16    = (f16*)  alloc((size_t)M_ * F_ * 2);
    f16*   u16    = (f16*)  alloc((size_t)M_ * F_ * 2);
    f16*   act16  = (f16*)  alloc((size_t)M_ * F_ * 2);
    float* basef  = (float*)alloc((size_t)M_ * D_ * 4);
    float* routf  = (float*)alloc((size_t)M_ * D_ * 4);

    auto cvt = [&](const float* in, f16* o, size_t n) {
        cvt_kernel<<<2048, 256, 0, stream>>>(in, o, n);
    };
    auto gemm = [&](const f16* A, const f16* Bm, void* C, int Md, int Nd, int Kd, int mode) {
        dim3 g(Nd / TILE_N, Md / TILE_M);
        gemm_wmma_kernel<<<g, 256, 0, stream>>>(A, Bm, C, Md, Nd, Kd, mode);
    };

    // 1. convert weights to f16 (we_gate/we_up transposed to [D, E*FE])
    cvt(wq, wq16, (size_t)D_ * 1024);
    cvt(wk, wk16, (size_t)D_ * 512);
    cvt(wv, wv16, (size_t)D_ * 512);
    cvt(wo, wo16, (size_t)1024 * D_);
    cvt(w_gate, wg16, (size_t)D_ * F_);
    cvt(w_up,   wu16, (size_t)D_ * F_);
    cvt(w_down, wd16, (size_t)F_ * D_);
    cvt(we_d,   wed16, (size_t)E_ * FE_ * D_);          // (E,FE,D) == [E*FE, D]
    cvt_texp_kernel<<<2048, 256, 0, stream>>>(we_g, weg16);
    cvt_texp_kernel<<<2048, 256, 0, stream>>>(we_u, weu16);

    // 2. h = rmsnorm(x) -> f16
    rmsnorm_kernel<<<M_, 256, 0, stream>>>(x, ln1_w, h16, D_);

    // 3. QKV projections (WMMA)
    gemm(h16, wq16, qf, M_, 1024, D_, 0);
    gemm(h16, wk16, kf, M_, 512,  D_, 0);
    gemm(h16, wv16, vf, M_, 512,  D_, 0);

    // 4. bias + RoPE + pack
    rope_pack_kernel<<<M_, 256, 0, stream>>>(qf, bq, q16, H_);
    rope_pack_kernel<<<M_, 256, 0, stream>>>(kf, bk, k16, KV_);
    pack_v_kernel<<<M_, 256, 0, stream>>>(vf, bv, v16t);

    // 5. flash attention (WMMA)
    attn_kernel<<<dim3(P_ / 16, H_, B_), 32, 0, stream>>>(q16, k16, v16t, attn16);

    // 6. output projection + residual + rmsnorm2
    gemm(attn16, wo16, of, M_, D_, 1024, 0);
    resid_rmsnorm_kernel<<<M_, 256, 0, stream>>>(x, of, ln2_w, x1, h2_16, D_);

    // 7. dense MLP: silu(h2@Wg) * (h2@Wu) @ Wd
    gemm(h2_16, wg16, g16, M_, F_, D_, 2);   // SiLU fused
    gemm(h2_16, wu16, u16, M_, F_, D_, 1);
    mul_kernel<<<8192, 256, 0, stream>>>(g16, u16, act16, (size_t)M_ * F_);
    gemm(act16, wd16, basef, M_, D_, F_, 0);

    // 8. routed experts as dense GEMMs (mask folded into activations)
    gemm(h2_16, weg16, g16, M_, F_, D_, 2);  // SiLU fused
    gemm(h2_16, weu16, u16, M_, F_, D_, 1);
    mask_mul_kernel<<<8192, 256, 0, stream>>>(g16, u16, emask, act16);
    gemm(act16, wed16, routf, M_, D_, F_, 0);

    // 9. out = x1 + base + routed
    final_add_kernel<<<8192, 256, 0, stream>>>(x1, basef, routf, out, (size_t)M_ * D_);
}
